// FlaxLLaMAAttention_83803401880279
// MI455X (gfx1250) — matrix operand order repaired and checked by reading, not compiled, not run on hardware
//
#include <hip/hip_runtime.h>

// ---------------------------------------------------------------------------
// LLaMA attention block for gfx1250 (MI455X): B=2, S=2048, D=2048, H=16, DH=128
// All heavy math on v_wmma_f32_16x16x32_bf16 (fp32 accumulate).
// Tile staging uses CDNA5 async global->LDS copies (ASYNCcnt) with LDS
// double-buffering in the GEMM, falling back to synchronous staging if the
// toolchain lacks the builtins.
// ---------------------------------------------------------------------------

#define BB   2
#define SS   2048
#define DD   2048
#define HH   16
#define DH   128
#define MROWS (BB * SS)        // 4096
#define NELE_X   (MROWS * DD)  // 8388608
#define NELE_W   (DD * DD)     // 4194304

#define AS1 __attribute__((address_space(1)))
#define AS3 __attribute__((address_space(3)))

typedef int v4i __attribute__((ext_vector_type(4)));

#if defined(__has_builtin)
#  if __has_builtin(__builtin_amdgcn_global_load_async_to_lds_b128)
#    define HAVE_ASYNC_COPY 1
#  else
#    define HAVE_ASYNC_COPY 0
#  endif
#  if __has_builtin(__builtin_amdgcn_s_wait_asynccnt)
#    define WAIT_ASYNC() __builtin_amdgcn_s_wait_asynccnt(0)
#  else
#    define WAIT_ASYNC() asm volatile("s_wait_asynccnt 0x0" ::: "memory")
#  endif
#else
#  define HAVE_ASYNC_COPY 0
#  define WAIT_ASYNC() ((void)0)
#endif

typedef __attribute__((ext_vector_type(16))) __bf16 v16bf;
typedef __attribute__((ext_vector_type(8)))  float  v8f;

union Frag16 { v16bf v; unsigned int u[8]; };

static __device__ __forceinline__ unsigned short f32_to_bf16(float f) {
    union { float f; unsigned int u; } c; c.f = f;
    unsigned int r = c.u + 0x7FFFu + ((c.u >> 16) & 1u);
    return (unsigned short)(r >> 16);
}
static __device__ __forceinline__ float bf16_to_f32(unsigned short h) {
    union { unsigned int u; float f; } c; c.u = ((unsigned int)h) << 16;
    return c.f;
}

static __device__ __forceinline__ v8f wmma_bf16(v16bf a, v16bf b, v8f c) {
    return __builtin_amdgcn_wmma_f32_16x16x32_bf16(
        /*neg_a=*/false, a, /*neg_b=*/false, b,
        /*c_mod=*/(short)0, c, /*reuse_a=*/false, /*reuse_b=*/false);
}

// 16B global -> LDS copy: async (no VGPR round-trip) when available.
static __device__ __forceinline__ void copy16_g2lds(const unsigned short* g,
                                                    unsigned short* l) {
#if HAVE_ASYNC_COPY
    __builtin_amdgcn_global_load_async_to_lds_b128((AS1 v4i*)g, (AS3 v4i*)l, 0, 0);
#else
    *(uint4*)l = *(const uint4*)g;
#endif
}

// ---------------------------------------------------------------------------
// fp32 -> bf16 conversion, 4 elements per thread
// ---------------------------------------------------------------------------
__global__ __launch_bounds__(256) void cvt_f32_to_bf16(const float* __restrict__ in,
                                                       unsigned short* __restrict__ out,
                                                       int n) {
    int i = (blockIdx.x * blockDim.x + threadIdx.x) * 4;
    if (i + 3 < n) {
        float4 v = *(const float4*)(in + i);
        ushort4 o;
        o.x = f32_to_bf16(v.x); o.y = f32_to_bf16(v.y);
        o.z = f32_to_bf16(v.z); o.w = f32_to_bf16(v.w);
        *(ushort4*)(out + i) = o;
    }
}

// ---------------------------------------------------------------------------
// WMMA GEMM: C[M,N] = A[M,K] * B[K,N], bf16 in, fp32 accum.
// Block tile 128x128, K-step 32, LDS double-buffered, async A staging.
// 256 threads = 8 waves as 2x4; each wave computes 64x32 = 4x2 fragments.
// ---------------------------------------------------------------------------
template <bool STORE_F32>
__global__ __launch_bounds__(256) void gemm_bf16_wmma(const unsigned short* __restrict__ A,
                                                      const unsigned short* __restrict__ B,
                                                      void* __restrict__ C,
                                                      int M, int N, int K) {
    __shared__ unsigned short Asm[2][128][40];   // [buf][m][k]
    __shared__ unsigned short Bsm[2][128][40];   // [buf][n][k] (transposed)

    const int t    = threadIdx.x;
    const int wave = t >> 5;
    const int lane = t & 31;
    const int grp  = lane >> 4;
    const int ln   = lane & 15;
    const int wr   = wave >> 2;
    const int wc   = wave & 3;
    const int m0   = blockIdx.y * 128;
    const int n0   = blockIdx.x * 128;

    v8f acc[4][2];
    #pragma unroll
    for (int i = 0; i < 4; ++i)
        #pragma unroll
        for (int j = 0; j < 2; ++j) acc[i][j] = {};

    auto stageA = [&](int kk, int buf) {
        for (int c = t; c < 512; c += 256) {             // 128 rows x 4 chunks
            int r = c >> 2, cc = c & 3;
            copy16_g2lds(A + (size_t)(m0 + r) * K + kk + cc * 8, &Asm[buf][r][cc * 8]);
        }
    };
    auto stageB = [&](int kk, int buf) {
        for (int c = t; c < 512; c += 256) {             // 32 rows x 16 chunks
            int r = c >> 4, cc = c & 15;
            uint4 d = *(const uint4*)(B + (size_t)(kk + r) * N + n0 + cc * 8);
            const unsigned short* e = (const unsigned short*)&d;
            #pragma unroll
            for (int i = 0; i < 8; ++i) Bsm[buf][cc * 8 + i][r] = e[i];
        }
    };

    stageA(0, 0);
    stageB(0, 0);
#if HAVE_ASYNC_COPY
    WAIT_ASYNC();
#endif
    __syncthreads();

    int cur = 0;
    for (int kk = 0; kk < K; kk += 32) {
        const int nxt = cur ^ 1;
        if (kk + 32 < K) {                    // stage next tile while computing
            stageA(kk + 32, nxt);
            stageB(kk + 32, nxt);
        }
        if (kk + 64 < K) {                    // warm GL2 for the tile after that
            __builtin_prefetch(A + (size_t)(m0 + (t >> 1)) * K + kk + 64, 0, 1);
            __builtin_prefetch(B + (size_t)(kk + 64 + (t >> 3)) * N + n0, 0, 1);
        }

        Frag16 af[4], bf[2];
        #pragma unroll
        for (int mf = 0; mf < 4; ++mf) {
            int mrow = wr * 64 + mf * 16 + ln;
            #pragma unroll
            for (int v = 0; v < 8; ++v) {
                int kx = (v >> 2) * 16 + grp * 8 + (v & 3) * 2;   // ISA 16b A layout
                af[mf].u[v] = *(const unsigned int*)&Asm[cur][mrow][kx];
            }
        }
        #pragma unroll
        for (int nf = 0; nf < 2; ++nf) {
            int ncol = wc * 32 + nf * 16 + ln;
            #pragma unroll
            for (int v = 0; v < 8; ++v) {
                int kx = (v >> 2) * 16 + grp * 8 + (v & 3) * 2;   // ISA 16b B layout
                bf[nf].u[v] = *(const unsigned int*)&Bsm[cur][ncol][kx];
            }
        }
        #pragma unroll
        for (int mf = 0; mf < 4; ++mf)
            #pragma unroll
            for (int nf = 0; nf < 2; ++nf)
                acc[mf][nf] = wmma_bf16(af[mf].v, bf[nf].v, acc[mf][nf]);

#if HAVE_ASYNC_COPY
        WAIT_ASYNC();
#endif
        __syncthreads();
        cur = nxt;
    }

    // Epilogue: C layout — VGPR r holds row M = 8*grp + r, col = ln
    #pragma unroll
    for (int mf = 0; mf < 4; ++mf) {
        #pragma unroll
        for (int nf = 0; nf < 2; ++nf) {
            #pragma unroll
            for (int r = 0; r < 8; ++r) {
                int row = m0 + wr * 64 + mf * 16 + grp * 8 + r;
                int col = n0 + wc * 32 + nf * 16 + ln;
                float val = acc[mf][nf][r];
                if (STORE_F32) ((float*)C)[(size_t)row * N + col] = val;
                else ((unsigned short*)C)[(size_t)row * N + col] = f32_to_bf16(val);
            }
        }
    }
}

// ---------------------------------------------------------------------------
// RoPE (+ [B,S,H,DH] -> [B,H,S,DH] transpose). apply==0: transpose only (V).
// ---------------------------------------------------------------------------
__global__ __launch_bounds__(256) void rope_transpose(const unsigned short* __restrict__ in,
                                                      unsigned short* __restrict__ out,
                                                      const int* __restrict__ pos_ids,
                                                      int apply) {
    int idx = blockIdx.x * blockDim.x + threadIdx.x;   // B*S*H*(DH/2) threads
    int k = idx & 63;
    int h = (idx >> 6) & 15;
    int s = (idx >> 10) & 2047;
    int b = idx >> 21;
    size_t iof = (((size_t)b * SS + s) * HH + h) * DH + 2 * k;
    size_t oof = (((size_t)b * HH + h) * SS + s) * DH + 2 * k;
    float a  = bf16_to_f32(in[iof]);
    float bb = bf16_to_f32(in[iof + 1]);
    if (apply) {
        const float L2T_OVER = 13.287712379549449f / 64.0f;   // log2(10000)/(DH/2)
        float pos = (float)pos_ids[b * SS + s];
        float ang = pos * exp2f(-(float)k * L2T_OVER);
        float sn, cs;
        __sincosf(ang, &sn, &cs);
        out[oof]     = f32_to_bf16(a * cs - bb * sn);
        out[oof + 1] = f32_to_bf16(a * sn + bb * cs);
    } else {
        out[oof]     = in[iof];
        out[oof + 1] = in[iof + 1];
    }
}

// ---------------------------------------------------------------------------
// Flash attention: Q/K/V in [B,H,S,DH] bf16 -> Ob [B,S,H*DH] bf16.
// Grid (S/64, B*H), 128 threads = 4 waves; each wave owns 16 query rows.
// Online softmax; 32-key tiles; async K staging; V transposed manually.
// ---------------------------------------------------------------------------
__global__ __launch_bounds__(128) void flash_attn(const unsigned short* __restrict__ Qh,
                                                  const unsigned short* __restrict__ Kh,
                                                  const unsigned short* __restrict__ Vh,
                                                  const int* __restrict__ amask,
                                                  unsigned short* __restrict__ Ob) {
    __shared__ unsigned short Ks[32][136];      // [key][dh]
    __shared__ unsigned short Vs[128][34];      // [dh][key] (transposed)
    __shared__ unsigned short Ps[4][16][34];    // per-wave P staging [qrow][key]

    const int bh   = blockIdx.y;
    const int b    = bh >> 4;
    const int qblk = blockIdx.x;
    const int wave = threadIdx.x >> 5;
    const int lane = threadIdx.x & 31;
    const int grp  = lane >> 4;
    const int ln   = lane & 15;
    const int qbase = qblk * 64 + wave * 16;
    const float scale = 0.08838834764831845f;   // 1/sqrt(128)

    const unsigned short* qptr = Qh + (size_t)bh * SS * DH;
    const unsigned short* kptr = Kh + (size_t)bh * SS * DH;
    const unsigned short* vptr = Vh + (size_t)bh * SS * DH;

    // Q fragments: 4 x (16x32) covering DH=128, rows qbase..qbase+15
    Frag16 qa[4];
    #pragma unroll
    for (int df = 0; df < 4; ++df)
        #pragma unroll
        for (int v = 0; v < 8; ++v) {
            int kx = df * 32 + (v >> 2) * 16 + grp * 8 + (v & 3) * 2;
            qa[df].u[v] = *(const unsigned int*)(qptr + (size_t)(qbase + ln) * DH + kx);
        }

    v8f o[8];
    #pragma unroll
    for (int f = 0; f < 8; ++f) o[f] = {};
    float mrow[8], lrow[8];
    #pragma unroll
    for (int r = 0; r < 8; ++r) { mrow[r] = -3.0e38f; lrow[r] = 0.0f; }

    const int nkt = 2 * qblk + 2;               // causal: keys <= qblk*64+63
    for (int it = 0; it < nkt; ++it) {
        const int kt = it * 32;
        // stage K tile [32 keys][128 dh] (async path when available)
        for (int c = threadIdx.x; c < 512; c += 128) {
            int r = c >> 4, cc = c & 15;
            copy16_g2lds(kptr + (size_t)(kt + r) * DH + cc * 8, &Ks[r][cc * 8]);
        }
        // stage V tile transposed -> [dh][key]
        for (int c = threadIdx.x; c < 512; c += 128) {
            int r = c >> 4, cc = c & 15;
            uint4 d = *(const uint4*)(vptr + (size_t)(kt + r) * DH + cc * 8);
            const unsigned short* e = (const unsigned short*)&d;
            #pragma unroll
            for (int i = 0; i < 8; ++i) Vs[cc * 8 + i][r] = e[i];
        }
#if HAVE_ASYNC_COPY
        WAIT_ASYNC();
#endif
        __syncthreads();

        // scores for 2 key sub-tiles of 16
        float st[2][8];
        #pragma unroll
        for (int nt = 0; nt < 2; ++nt) {
            v8f sacc = {};
            #pragma unroll
            for (int df = 0; df < 4; ++df) {
                Frag16 kf;
                #pragma unroll
                for (int v = 0; v < 8; ++v)
                    kf.u[v] = *(const unsigned int*)&Ks[nt * 16 + ln][df * 32 + (v >> 2) * 16 + grp * 8 + (v & 3) * 2];
                sacc = wmma_bf16(qa[df].v, kf.v, sacc);
            }
            int keyi = kt + nt * 16 + ln;
            int mv   = amask[b * SS + keyi];
            #pragma unroll
            for (int r = 0; r < 8; ++r) {
                int q = qbase + grp * 8 + r;
                float sv = sacc[r] * scale;
                if (keyi > q || mv == 0) sv = -3.0e38f;
                st[nt][r] = sv;
            }
        }

        // online softmax (row reduce across the 16-lane group)
        float mnew[8], alpha[8];
        #pragma unroll
        for (int r = 0; r < 8; ++r) {
            float mx = fmaxf(st[0][r], st[1][r]);
            #pragma unroll
            for (int off = 1; off < 16; off <<= 1)
                mx = fmaxf(mx, __shfl_xor(mx, off, 32));
            mnew[r]  = fmaxf(mrow[r], mx);
            alpha[r] = __expf(mrow[r] - mnew[r]);
            mrow[r]  = mnew[r];
        }
        #pragma unroll
        for (int r = 0; r < 8; ++r) {
            float p0 = __expf(st[0][r] - mnew[r]);
            float p1 = __expf(st[1][r] - mnew[r]);
            int qr = grp * 8 + r;
            Ps[wave][qr][ln]      = f32_to_bf16(p0);
            Ps[wave][qr][16 + ln] = f32_to_bf16(p1);
            float ps = p0 + p1;
            #pragma unroll
            for (int off = 1; off < 16; off <<= 1)
                ps += __shfl_xor(ps, off, 32);
            lrow[r] = alpha[r] * lrow[r] + ps;
        }
        // rescale running output
        #pragma unroll
        for (int f = 0; f < 8; ++f)
            #pragma unroll
            for (int r = 0; r < 8; ++r) o[f][r] *= alpha[r];

        // P as A-fragment (16x32) from per-wave LDS staging
        Frag16 pf;
        #pragma unroll
        for (int v = 0; v < 8; ++v) {
            int kx = (v >> 2) * 16 + grp * 8 + (v & 3) * 2;
            pf.u[v] = *(const unsigned int*)&Ps[wave][ln][kx];
        }
        // O += P * V
        #pragma unroll
        for (int nf = 0; nf < 8; ++nf) {
            Frag16 vf;
            #pragma unroll
            for (int v = 0; v < 8; ++v)
                vf.u[v] = *(const unsigned int*)&Vs[nf * 16 + ln][(v >> 2) * 16 + grp * 8 + (v & 3) * 2];
            o[nf] = wmma_bf16(pf.v, vf.v, o[nf]);
        }
        __syncthreads();
    }

    // epilogue: write [B,S,H*DH] bf16
    const int h = bh & 15;
    #pragma unroll
    for (int nf = 0; nf < 8; ++nf)
        #pragma unroll
        for (int r = 0; r < 8; ++r) {
            int q  = qbase + grp * 8 + r;
            int dh = nf * 16 + ln;
            float lr = lrow[r] > 0.0f ? lrow[r] : 1.0f;
            Ob[((size_t)b * SS + q) * DD + h * DH + dh] = f32_to_bf16(o[nf][r] / lr);
        }
}

// ---------------------------------------------------------------------------
// Launcher
// ---------------------------------------------------------------------------
extern "C" void kernel_launch(void* const* d_in, const int* in_sizes, int n_in,
                              void* d_out, int out_size, void* d_ws, size_t ws_size,
                              hipStream_t stream) {
    const float* hs = (const float*)d_in[0];
    const float* wq = (const float*)d_in[1];
    const float* wk = (const float*)d_in[2];
    const float* wv = (const float*)d_in[3];
    const float* wo = (const float*)d_in[4];
    const int* amask = (const int*)d_in[5];
    const int* pos   = (const int*)d_in[6];
    float* out = (float*)d_out;

    unsigned short* p = (unsigned short*)d_ws;
    unsigned short* Xb  = p; p += NELE_X;   // hidden bf16
    unsigned short* Wqb = p; p += NELE_W;
    unsigned short* Wkb = p; p += NELE_W;
    unsigned short* Wvb = p; p += NELE_W;
    unsigned short* Wob = p; p += NELE_W;
    unsigned short* Qb  = p; p += NELE_X;   // [B,S,D] pre-rope
    unsigned short* Kb  = p; p += NELE_X;
    unsigned short* Vb  = p; p += NELE_X;
    unsigned short* Qh  = p; p += NELE_X;   // [B,H,S,DH]
    unsigned short* Kh  = p; p += NELE_X;
    unsigned short* Vh  = p; p += NELE_X;
    unsigned short* Ob  = p; p += NELE_X;   // attention out [B,S,D]

    // 1. conversions
    cvt_f32_to_bf16<<<NELE_X / 4 / 256, 256, 0, stream>>>(hs, Xb, NELE_X);
    cvt_f32_to_bf16<<<NELE_W / 4 / 256, 256, 0, stream>>>(wq, Wqb, NELE_W);
    cvt_f32_to_bf16<<<NELE_W / 4 / 256, 256, 0, stream>>>(wk, Wkb, NELE_W);
    cvt_f32_to_bf16<<<NELE_W / 4 / 256, 256, 0, stream>>>(wv, Wvb, NELE_W);
    cvt_f32_to_bf16<<<NELE_W / 4 / 256, 256, 0, stream>>>(wo, Wob, NELE_W);

    // 2. QKV projections
    dim3 ggrid(DD / 128, MROWS / 128);
    gemm_bf16_wmma<false><<<ggrid, 256, 0, stream>>>(Xb, Wqb, Qb, MROWS, DD, DD);
    gemm_bf16_wmma<false><<<ggrid, 256, 0, stream>>>(Xb, Wkb, Kb, MROWS, DD, DD);
    gemm_bf16_wmma<false><<<ggrid, 256, 0, stream>>>(Xb, Wvb, Vb, MROWS, DD, DD);

    // 3. RoPE + head-major transpose
    int npairs = BB * SS * HH * (DH / 2);   // 4194304
    rope_transpose<<<npairs / 256, 256, 0, stream>>>(Qb, Qh, pos, 1);
    rope_transpose<<<npairs / 256, 256, 0, stream>>>(Kb, Kh, pos, 1);
    rope_transpose<<<npairs / 256, 256, 0, stream>>>(Vb, Vh, pos, 0);

    // 4. flash attention
    flash_attn<<<dim3(SS / 64, BB * HH), 128, 0, stream>>>(Qh, Kh, Vh, amask, Ob);

    // 5. output projection (fp32 epilogue straight to d_out)
    gemm_bf16_wmma<true><<<ggrid, 256, 0, stream>>>(Ob, Wob, out, MROWS, DD, DD);
}
